// SynthesisBlock_73830487818666
// MI455X (gfx1250) — matrix operand order repaired and checked by reading, not compiled, not run on hardware
//
#include <hip/hip_runtime.h>

#define B_      2
#define F_C     32768
#define F_F     131072
#define C_IN    128
#define C_OUT   64
#define W_DIM   512
#define COLOR   3
#define K2      9
#define NROWS1  (B_*F_C)    // 65536
#define NROWS2  (B_*F_F)    // 262144
#define KS0     36          // 1152/32 total k-steps conv1
#define KS1     18          // 576/32  total k-steps conv2

typedef __attribute__((ext_vector_type(16))) __bf16 v16bf;
typedef __attribute__((ext_vector_type(8)))  float  v8f;
typedef __attribute__((ext_vector_type(8)))  unsigned short u16x8;

union ABfrag { unsigned short s[16]; u16x8 h[2]; v16bf v; };

__device__ __forceinline__ unsigned short f2bf(float f) {
  unsigned int u = __builtin_bit_cast(unsigned int, f);
  unsigned int r = u + 0x7FFFu + ((u >> 16) & 1u);   // round-to-nearest-even
  return (unsigned short)(r >> 16);
}

// ---------------------------------------------------------------- styles
__global__ void prep_styles(const float* __restrict__ wsIn,
                            const float* __restrict__ a0w, const float* __restrict__ a0b,
                            const float* __restrict__ a1w, const float* __restrict__ a1b,
                            const float* __restrict__ atw, const float* __restrict__ atb,
                            const float* __restrict__ Wt,
                            float* __restrict__ s0, float* __restrict__ s1,
                            float* __restrict__ st, float* __restrict__ wtmod) {
  const int tid = threadIdx.x;                  // 512 threads
  const float invs = 0.044194173824159216f;     // 1/sqrt(512)
  if (tid < 256) {
    int b = tid >> 7, i = tid & 127;
    const float* w  = wsIn + (b*3 + 0)*W_DIM;
    const float* aw = a0w + i*W_DIM;
    float acc = 0.f;
    for (int j = 0; j < W_DIM; ++j) acc += w[j]*aw[j];
    s0[b*C_IN + i] = acc*invs + a0b[i];
  } else if (tid < 384) {
    int t = tid - 256; int b = t >> 6, i = t & 63;
    const float* w  = wsIn + (b*3 + 1)*W_DIM;
    const float* aw = a1w + i*W_DIM;
    float acc = 0.f;
    for (int j = 0; j < W_DIM; ++j) acc += w[j]*aw[j];
    s1[b*C_OUT + i] = acc*invs + a1b[i];
  } else {
    int t = tid - 384; int b = t >> 6, i = t & 63;
    const float* w  = wsIn + (b*3 + 2)*W_DIM;
    const float* aw = atw + i*W_DIM;
    float acc = 0.f;
    for (int j = 0; j < W_DIM; ++j) acc += w[j]*aw[j];
    st[b*C_OUT + i] = (acc*invs + atb[i]) * 0.125f;   // fold 1/sqrt(C_OUT)
  }
  __syncthreads();
  if (tid < B_*COLOR*C_OUT) {                   // 384: Wtmod[b][c][i]=Wt[c][i]*st[b][i]
    int b = tid / (COLOR*C_OUT);
    int rem = tid % (COLOR*C_OUT);
    int c = rem / C_OUT, i = rem % C_OUT;
    wtmod[tid] = Wt[c*C_OUT + i] * st[b*C_OUT + i];
  }
}

// ---------------------------------------------------------------- demod factors
__global__ void prep_demod(const float* __restrict__ W0, const float* __restrict__ s0,
                           const float* __restrict__ W1, const float* __restrict__ s1,
                           float* __restrict__ d0, float* __restrict__ d1) {
  __shared__ float red[128];
  const int blk = blockIdx.x;     // 0..255 : 128 pairs for W0, 128 for W1
  const int tid = threadIdx.x;    // 128
  float acc = 0.f;
  if (blk < 128) {
    int b = blk >> 6, o = blk & 63;
    const float* wrow = W0 + (size_t)o*(C_IN*K2);
    const float* s    = s0 + b*C_IN;
    for (int t = tid; t < C_IN*K2; t += 128) {
      float w = wrow[t] * s[t / K2];
      acc += w*w;
    }
  } else {
    int pr = blk - 128; int b = pr >> 6, o = pr & 63;
    const float* wrow = W1 + (size_t)o*(C_OUT*K2);
    const float* s    = s1 + b*C_OUT;
    for (int t = tid; t < C_OUT*K2; t += 128) {
      float w = wrow[t] * s[t / K2];
      acc += w*w;
    }
  }
  red[tid] = acc;
  __syncthreads();
  for (int off = 64; off > 0; off >>= 1) {
    if (tid < off) red[tid] += red[tid + off];
    __syncthreads();
  }
  if (tid == 0) {
    float d = rsqrtf(red[0] + 1e-8f);
    if (blk < 128) d0[blk] = d; else d1[blk - 128] = d;
  }
}

// -------------------------------------------- pack demodulated weights as WMMA B fragments
// layout: ((b*4+nt)*TOTKS + gks)*512 + lane*16 + t ;  K = gks*32 + (lane>>4)*16 + t ; N = nt*16+(lane&15)
__global__ void prep_pack(const float* __restrict__ W0, const float* __restrict__ s0,
                          const float* __restrict__ d0,
                          const float* __restrict__ W1, const float* __restrict__ s1,
                          const float* __restrict__ d1,
                          unsigned short* __restrict__ w0f, unsigned short* __restrict__ w1f) {
  const int e = blockIdx.x*blockDim.x + threadIdx.x;
  const int N0 = B_*4*KS0*512;   // 147456
  const int N1 = B_*4*KS1*512;   // 73728
  if (e < N0) {
    int t = e & 15, lane = (e >> 4) & 31;
    int gks = (e >> 9) % KS0;
    int nt  = (e / (512*KS0)) & 3;
    int b   = e / (512*KS0*4);
    int kap = gks*32 + (lane >> 4)*16 + t;     // flat K index = kk*C_IN + i
    int i = kap % C_IN, kk = kap / C_IN;
    int o = nt*16 + (lane & 15);
    float v = W0[((size_t)o*C_IN + i)*K2 + kk] * s0[b*C_IN + i] * d0[b*C_OUT + o];
    w0f[e] = f2bf(v);
  } else if (e < N0 + N1) {
    int e1 = e - N0;
    int t = e1 & 15, lane = (e1 >> 4) & 31;
    int gks = (e1 >> 9) % KS1;
    int nt  = (e1 / (512*KS1)) & 3;
    int b   = e1 / (512*KS1*4);
    int kap = gks*32 + (lane >> 4)*16 + t;     // flat K index = kk*C_OUT + i
    int i = kap % C_OUT, kk = kap / C_OUT;
    int o = nt*16 + (lane & 15);
    float v = W1[((size_t)o*C_OUT + i)*K2 + kk] * s1[b*C_OUT + i] * d1[b*C_OUT + o];
    w1f[e1] = f2bf(v);
  }
}

// ---------------------------------------------------------------- cast x -> bf16
__global__ void cast_x(const float* __restrict__ x, unsigned short* __restrict__ xbf, int n) {
  int t = blockIdx.x*blockDim.x + threadIdx.x;
  if (t < n) xbf[t] = f2bf(x[t]);
}

// ---------------------------------------------------------------- resolve gather indices
__global__ void resolve_idx(const int* __restrict__ fn0, const int* __restrict__ fip0,
                            const int* __restrict__ fn1, const int* __restrict__ fip1,
                            const int* __restrict__ pool,
                            int* __restrict__ rn0, int* __restrict__ rn1,
                            int* __restrict__ pu1, float* __restrict__ invcnt1) {
  int t = blockIdx.x*blockDim.x + threadIdx.x;
  if (t < NROWS2) {
    int cnt = 0;
    #pragma unroll
    for (int k = 0; k < K2; ++k) {
      int f = fn1[t*K2 + k];
      int p = fip1[t*K2 + k];
      cnt += (p == 0);
      int v = (p != 0 || f >= NROWS2) ? -1 : f;
      rn1[t*K2 + k] = v;
      pu1[t*K2 + k] = (v >= 0) ? pool[v] : -1;
    }
    invcnt1[t] = 1.0f / (float)(cnt > 0 ? cnt : 1);
  } else if (t < NROWS2 + NROWS1) {
    int r = t - NROWS2;
    #pragma unroll
    for (int k = 0; k < K2; ++k) {
      int f = fn0[r*K2 + k];
      int p = fip0[r*K2 + k];
      rn0[r*K2 + k] = (p != 0 || f >= NROWS1) ? -1 : f;
    }
  }
}

// ---------------------------------------------------------------- gathered GEMM via WMMA bf16
// One wave = one 16-row output tile; block = 4 waves = 64 rows. K = 9 neighbors * CNBR channels.
template<int CNBR, bool EPI>
__global__ void __launch_bounds__(128)
modconv_gemm(const unsigned short* __restrict__ Asrc,   // bf16 source rows [*][CNBR]
             const int* __restrict__ ridx,              // [M][9] resolved (-1 = zero)
             const unsigned short* __restrict__ Bfrag,  // packed weight fragments
             float* __restrict__ Out,
             const float* __restrict__ bias,
             const float* __restrict__ nc,
             const float* __restrict__ ns,
             int rowsPerBatch) {
  constexpr int KSTEPS = CNBR / 32;
  constexpr int TOTKS  = K2 * KSTEPS;
  __shared__ unsigned short lds[4][16][CNBR + 8];       // +16B pad destaggers banks
  const int wave = threadIdx.x >> 5;
  const int lane = threadIdx.x & 31;
  const int row0 = (blockIdx.x*4 + wave) * 16;
  const int batch = row0 / rowsPerBatch;
  const int hs = lane >> 4, m = lane & 15;
  const int gr = lane >> 1, gh = lane & 1;              // gather: 2 lanes per row
  v8f acc[4] = {};
  const unsigned short* bbase = Bfrag + (size_t)batch*4*TOTKS*512 + lane*16;

  for (int nbr = 0; nbr < K2; ++nbr) {
    // ---- gather 16 rows x CNBR bf16 into LDS (zeros for pad/masked) ----
    int src = ridx[(row0 + gr)*K2 + nbr];
    const unsigned short* sp = (src >= 0) ? (Asrc + (size_t)src*CNBR + gh*(CNBR/2)) : nullptr;
    unsigned short* dst = &lds[wave][gr][gh*(CNBR/2)];
    #pragma unroll
    for (int j = 0; j < CNBR/2; j += 8) {
      u16x8 v = {0,0,0,0,0,0,0,0};
      if (sp) v = *(const u16x8*)(sp + j);
      *(u16x8*)(dst + j) = v;
    }
    __syncthreads();
    // ---- WMMA over this neighbor's K chunk ----
    #pragma unroll
    for (int ks = 0; ks < KSTEPS; ++ks) {
      ABfrag a;
      const unsigned short* lp = &lds[wave][m][ks*32 + hs*16];
      a.h[0] = *(const u16x8*)lp;           // local K = hs*16   .. +7
      a.h[1] = *(const u16x8*)(lp + 8);     // local K = hs*16+8 .. +15
      const int gks = nbr*KSTEPS + ks;
      #pragma unroll
      for (int nt = 0; nt < 4; ++nt) {
        ABfrag bb;
        const unsigned short* bp = bbase + (size_t)(nt*TOTKS + gks)*512;
        bb.h[0] = *(const u16x8*)bp;
        bb.h[1] = *(const u16x8*)(bp + 8);
        acc[nt] = __builtin_amdgcn_wmma_f32_16x16x32_bf16(
            false, a.v, false, bb.v, (short)0, acc[nt], false, false);
      }
    }
    __syncthreads();
  }

  // ---- epilogue / store (C/D layout: lane: N=m, rows hs*8+v) ----
  const float nss = EPI ? ns[0] : 0.f;
  #pragma unroll
  for (int nt = 0; nt < 4; ++nt) {
    #pragma unroll
    for (int v = 0; v < 8; ++v) {
      const int rowg = row0 + hs*8 + v;
      const int col  = nt*16 + m;
      float val = acc[nt][v];
      if (EPI) {
        val += nc[rowg & (F_F - 1)]*nss + bias[col];
        val = (val >= 0.f) ? val : 0.2f*val;
        val *= 1.4142135623730951f;
        val = fminf(fmaxf(val, -256.f), 256.f);
      }
      Out[(size_t)rowg*C_OUT + col] = val;
    }
  }
}

// ---------------------------------------------------------------- upsample + bias + act -> bf16
__global__ void upsample_act(const float* __restrict__ h1,
                             const int* __restrict__ pu1,
                             const float* __restrict__ invcnt1,
                             const float* __restrict__ nc0, const float* __restrict__ ns0,
                             const float* __restrict__ b0,
                             unsigned short* __restrict__ hact) {
  int t = blockIdx.x*blockDim.x + threadIdx.x;   // NROWS2 * 16 threads (4 ch each)
  int r = t >> 4, q = t & 15;
  if (r >= NROWS2) return;
  float a0 = 0.f, a1 = 0.f, a2 = 0.f, a3 = 0.f;
  #pragma unroll
  for (int k = 0; k < K2; ++k) {
    int p = pu1[r*K2 + k];
    if (p >= 0) {
      const float* hp = h1 + (size_t)p*C_OUT + q*4;
      a0 += hp[0]; a1 += hp[1]; a2 += hp[2]; a3 += hp[3];
    }
  }
  const float ic = invcnt1[r];
  const float noise = nc0[r & (F_F - 1)] * ns0[0];
  float vals[4] = {a0, a1, a2, a3};
  #pragma unroll
  for (int c = 0; c < 4; ++c) {
    float val = vals[c]*ic + noise + b0[q*4 + c];
    val = (val >= 0.f) ? val : 0.2f*val;
    val *= 1.4142135623730951f;
    val = fminf(fmaxf(val, -256.f), 256.f);
    hact[(size_t)r*C_OUT + q*4 + c] = f2bf(val);
  }
}

// ---------------------------------------------------------------- toRGB + image upsample
__global__ void torgb_img(const float* __restrict__ hout,
                          const float* __restrict__ img,
                          const int* __restrict__ rn1,
                          const int* __restrict__ pu1,
                          const float* __restrict__ invcnt1,
                          const float* __restrict__ wtmod,
                          const float* __restrict__ bt,
                          float* __restrict__ imgOut) {
  int r = blockIdx.x*blockDim.x + threadIdx.x;
  if (r >= NROWS2) return;
  const int b = r / F_F;
  float y0 = 0.f, y1 = 0.f, y2 = 0.f;
  const int i0 = rn1[r*K2];
  if (i0 >= 0) {
    const float* hp = hout + (size_t)i0*C_OUT;
    const float* w0 = wtmod + (size_t)b*COLOR*C_OUT;
    const float* w1 = w0 + C_OUT;
    const float* w2 = w1 + C_OUT;
    for (int i = 0; i < C_OUT; ++i) {
      float h = hp[i];
      y0 += h*w0[i]; y1 += h*w1[i]; y2 += h*w2[i];
    }
  }
  y0 = fminf(fmaxf(y0 + bt[0], -256.f), 256.f);
  y1 = fminf(fmaxf(y1 + bt[1], -256.f), 256.f);
  y2 = fminf(fmaxf(y2 + bt[2], -256.f), 256.f);
  float m0 = 0.f, m1 = 0.f, m2 = 0.f;
  #pragma unroll
  for (int k = 0; k < K2; ++k) {
    int p = pu1[r*K2 + k];
    if (p >= 0) {
      const float* ip = img + (size_t)p*COLOR;
      m0 += ip[0]; m1 += ip[1]; m2 += ip[2];
    }
  }
  const float ic = invcnt1[r];
  imgOut[(size_t)r*3 + 0] = m0*ic + y0;
  imgOut[(size_t)r*3 + 1] = m1*ic + y1;
  imgOut[(size_t)r*3 + 2] = m2*ic + y2;
}

// ----------------------------------------------------------------
extern "C" void kernel_launch(void* const* d_in, const int* in_sizes, int n_in,
                              void* d_out, int out_size, void* d_ws, size_t ws_size,
                              hipStream_t stream) {
  (void)in_sizes; (void)n_in; (void)out_size; (void)ws_size;
  const float* x    = (const float*)d_in[0];
  const float* img  = (const float*)d_in[1];
  const float* wsv  = (const float*)d_in[2];
  const float* a0w  = (const float*)d_in[3];
  const float* a0b  = (const float*)d_in[4];
  const float* W0   = (const float*)d_in[5];
  const float* b0   = (const float*)d_in[6];
  const float* ns0  = (const float*)d_in[7];
  const float* nc0  = (const float*)d_in[8];
  const float* a1w  = (const float*)d_in[9];
  const float* a1b  = (const float*)d_in[10];
  const float* W1   = (const float*)d_in[11];
  const float* b1   = (const float*)d_in[12];
  const float* ns1  = (const float*)d_in[13];
  const float* nc1  = (const float*)d_in[14];
  const float* atw  = (const float*)d_in[15];
  const float* atb  = (const float*)d_in[16];
  const float* Wt   = (const float*)d_in[17];
  const float* bt   = (const float*)d_in[18];
  const int*   fn0  = (const int*)d_in[19];
  const int*   fip0 = (const int*)d_in[20];
  const int*   fn1  = (const int*)d_in[21];
  const int*   fip1 = (const int*)d_in[22];
  const int*   pool = (const int*)d_in[23];

  char* p = (char*)d_ws;
  auto alloc = [&](size_t bytes) -> void* {
    void* r = (void*)p;
    p += (bytes + 255) & ~(size_t)255;
    return r;
  };
  float* s0            = (float*)alloc((size_t)B_*C_IN*4);
  float* s1v           = (float*)alloc((size_t)B_*C_OUT*4);
  float* stv           = (float*)alloc((size_t)B_*C_OUT*4);
  float* wtm           = (float*)alloc((size_t)B_*COLOR*C_OUT*4);
  float* d0            = (float*)alloc((size_t)B_*C_OUT*4);
  float* d1            = (float*)alloc((size_t)B_*C_OUT*4);
  unsigned short* w0f  = (unsigned short*)alloc((size_t)B_*4*KS0*512*2);
  unsigned short* w1f  = (unsigned short*)alloc((size_t)B_*4*KS1*512*2);
  unsigned short* xbf  = (unsigned short*)alloc((size_t)NROWS1*C_IN*2);
  float* h1            = (float*)alloc((size_t)NROWS1*C_OUT*4);
  unsigned short* hact = (unsigned short*)alloc((size_t)NROWS2*C_OUT*2);
  int* rn0             = (int*)alloc((size_t)NROWS1*K2*4);
  int* rn1             = (int*)alloc((size_t)NROWS2*K2*4);
  int* pu1             = (int*)alloc((size_t)NROWS2*K2*4);
  float* icnt          = (float*)alloc((size_t)NROWS2*4);

  float* hout = (float*)d_out;
  float* iout = hout + (size_t)NROWS2*C_OUT;

  prep_styles<<<1, 512, 0, stream>>>(wsv, a0w, a0b, a1w, a1b, atw, atb, Wt, s0, s1v, stv, wtm);
  prep_demod<<<256, 128, 0, stream>>>(W0, s0, W1, s1v, d0, d1);
  prep_pack<<<(B_*4*KS0*512 + B_*4*KS1*512 + 255)/256, 256, 0, stream>>>(W0, s0, d0, W1, s1v, d1, w0f, w1f);
  cast_x<<<(NROWS1*C_IN)/256, 256, 0, stream>>>(x, xbf, NROWS1*C_IN);
  resolve_idx<<<(NROWS2 + NROWS1 + 255)/256, 256, 0, stream>>>(fn0, fip0, fn1, fip1, pool, rn0, rn1, pu1, icnt);

  modconv_gemm<C_IN, false><<<NROWS1/64, 128, 0, stream>>>(xbf, rn0, w0f, h1, nullptr, nullptr, nullptr, F_C);
  upsample_act<<<(NROWS2*16)/256, 256, 0, stream>>>(h1, pu1, icnt, nc0, ns0, b0, hact);
  modconv_gemm<C_OUT, true><<<NROWS2/64, 128, 0, stream>>>(hact, rn1, w1f, hout, b1, nc1, ns1, F_F);
  torgb_img<<<NROWS2/256, 256, 0, stream>>>(hout, img, rn1, pu1, icnt, wtm, bt, iout);
}
